// Attention_10307921510428
// MI455X (gfx1250) — matrix operand order repaired and checked by reading, not compiled, not run on hardware
//
#include <hip/hip_runtime.h>

typedef __attribute__((ext_vector_type(16))) _Float16 v16h;
typedef __attribute__((ext_vector_type(8)))  float    v8f;
typedef __attribute__((ext_vector_type(4)))  int      v4i;

// ---------------- constants ----------------
#define Bb   8
#define Nn   1024
#define Cc   768
#define Hh   12
#define Dd   64
#define MTOT (Bb * Nn)          // 8192 rows
#define QKV_O (3 * Cc)          // 2304

// workspace layout (in _Float16 elements)
#define OFF_XH     ((size_t)0)
#define OFF_WQKV   ((size_t)(MTOT * Cc))
#define OFF_WPROJ  (OFF_WQKV + (size_t)(QKV_O * Cc))
#define OFF_Q      (OFF_WPROJ + (size_t)(Cc * Cc))
#define OFF_K      (OFF_Q + (size_t)(Bb * Hh * Nn * Dd))
#define OFF_V      (OFF_K + (size_t)(Bb * Hh * Nn * Dd))
#define OFF_AO     (OFF_V + (size_t)(Bb * Hh * Nn * Dd))

// ---------------- CDNA5 async copy to LDS (ASYNCcnt path) ----------------
#if __has_builtin(__builtin_amdgcn_global_load_async_to_lds_b128)
#define ASYNC_BUILTIN 1
#endif

typedef __attribute__((address_space(1))) v4i glob_v4i;
typedef __attribute__((address_space(3))) v4i lds_v4i;

__device__ __forceinline__ void async_copy16(void* lds, const void* glb) {
#if defined(ASYNC_BUILTIN)
  __builtin_amdgcn_global_load_async_to_lds_b128(
      (glob_v4i*)glb, (lds_v4i*)lds, 0, 0);
#else
  uint32_t l = (uint32_t)(uintptr_t)(lds_v4i*)lds;
  asm volatile("global_load_async_to_lds_b128 %0, %1, off"
               :: "v"(l), "v"(glb) : "memory");
#endif
}

__device__ __forceinline__ void wait_async0() {
#if __has_builtin(__builtin_amdgcn_s_wait_asynccnt)
  __builtin_amdgcn_s_wait_asynccnt(0);
#else
  asm volatile("s_wait_asynccnt 0x0" ::: "memory");
#endif
}

__device__ __forceinline__ void wait_ds0() {
#if __has_builtin(__builtin_amdgcn_s_wait_dscnt)
  __builtin_amdgcn_s_wait_dscnt(0);
#else
  asm volatile("s_wait_dscnt 0x0" ::: "memory");
#endif
}

// ---------------- fragment loaders (row-major f16 slab) ----------------
__device__ __forceinline__ v16h load_frag_a(const _Float16* base, int pitch,
                                            int rowBase, int kc, int lane) {
  const _Float16* p = base + (size_t)(rowBase + (lane & 15)) * pitch
                           + kc + ((lane >> 4) << 3);
  union { v16h v; uint4 u[2]; } r;
  r.u[0] = *(const uint4*)p;
  r.u[1] = *(const uint4*)(p + 16);
  return r.v;
}

__device__ __forceinline__ v16h load_frag_b(const _Float16* base, int pitch,
                                            int colBase, int kc, int lane) {
  const _Float16* p = base + (size_t)(colBase + (lane & 15)) * pitch
                           + kc + ((lane >> 4) << 3);
  union { v16h v; uint4 u[2]; } r;
  r.u[0] = *(const uint4*)p;
  r.u[1] = *(const uint4*)(p + 16);
  return r.v;
}

__device__ __forceinline__ v8f wmma_f16(v16h a, v16h b, v8f c) {
  return __builtin_amdgcn_wmma_f32_16x16x32_f16(false, a, false, b,
                                                (short)0, c, false, false);
}

// ---------------- f32 -> f16 convert (8 elements / thread) ----------------
__global__ __launch_bounds__(256) void cvt_f16(const float* __restrict__ in,
                                               _Float16* __restrict__ out, int n8) {
  int i = blockIdx.x * 256 + threadIdx.x;
  if (i < n8) {
    const float4* p = (const float4*)in + (size_t)i * 2;
    float4 a = p[0], b = p[1];
    union { _Float16 h[8]; uint4 u; } r;
    r.h[0] = (_Float16)a.x; r.h[1] = (_Float16)a.y;
    r.h[2] = (_Float16)a.z; r.h[3] = (_Float16)a.w;
    r.h[4] = (_Float16)b.x; r.h[5] = (_Float16)b.y;
    r.h[6] = (_Float16)b.z; r.h[7] = (_Float16)b.w;
    *((uint4*)(out + (size_t)i * 8)) = r.u;
  }
}

// ---------------- shared 768-K GEMM mainloop ----------------
// block = 8 waves computes a 64x128 tile; wave tile = 16 rows x 64 cols (4 WMMA n-subtiles)
// double-buffered LDS, async global->LDS staging
__device__ __forceinline__ void mm768x4(const _Float16* __restrict__ A,
                                        const _Float16* __restrict__ W,
                                        int m0, int n0,
                                        _Float16 (*As)[64 * 64],
                                        _Float16 (*Bs)[128 * 64],
                                        int t, v8f acc[4]) {
  const int lane = t & 31, wave = t >> 5;
  const int wr = (wave >> 1) << 4;   // 0/16/32/48
  const int wc = (wave & 1) << 6;    // 0/64
#pragma unroll
  for (int f = 0; f < 4; ++f)
#pragma unroll
    for (int r = 0; r < 8; ++r) acc[f][r] = 0.0f;

  auto stage = [&](int buf, int kc0) {
#pragma unroll
    for (int j = 0; j < 2; ++j) {   // A: 64x64 halves
      int idx = t * 16 + j * 8;
      async_copy16(&As[buf][idx],
                   &A[(size_t)(m0 + (idx >> 6)) * Cc + kc0 + (idx & 63)]);
    }
#pragma unroll
    for (int j = 0; j < 4; ++j) {   // W: 128x64 halves
      int idx = t * 32 + j * 8;
      async_copy16(&Bs[buf][idx],
                   &W[(size_t)(n0 + (idx >> 6)) * Cc + kc0 + (idx & 63)]);
    }
  };

  stage(0, 0);
  wait_async0();
  __syncthreads();
  for (int i = 0; i < 12; ++i) {
    if (i + 1 < 12) stage((i + 1) & 1, (i + 1) * 64);
    const _Float16* Ab = As[i & 1];
    const _Float16* Bb2 = Bs[i & 1];
#pragma unroll
    for (int ks = 0; ks < 64; ks += 32) {
      v16h a = load_frag_a(Ab, 64, wr, ks, lane);
#pragma unroll
      for (int f = 0; f < 4; ++f) {
        v16h b = load_frag_b(Bb2, 64, wc + f * 16, ks, lane);
        acc[f] = wmma_f16(a, b, acc[f]);
      }
    }
    wait_async0();
    __syncthreads();
  }
}

// ---------------- QKV GEMM: qkv = x @ w_qkv^T, scatter into q / k / vT ----------------
__global__ __launch_bounds__(256) void gemm_qkv(const _Float16* __restrict__ xh,
                                                const _Float16* __restrict__ wh,
                                                _Float16* __restrict__ qh,
                                                _Float16* __restrict__ kh,
                                                _Float16* __restrict__ vTh) {
  __shared__ __align__(16) _Float16 As[2][64 * 64];
  __shared__ __align__(16) _Float16 Bs[2][128 * 64];
  const int t = threadIdx.x;
  const int m0 = blockIdx.y * 64;
  const int n0 = blockIdx.x * 128;
  v8f acc[4];
  mm768x4(xh, wh, m0, n0, As, Bs, t, acc);

  const int lane = t & 31, wave = t >> 5;
  const int wr = (wave >> 1) << 4, wc = (wave & 1) << 6;
  const int mbase = m0 + wr + ((lane >> 4) << 3);
#pragma unroll
  for (int f = 0; f < 4; ++f) {
    const int oc = n0 + wc + f * 16 + (lane & 15);
    const int which = oc / Cc;
    const int rem = oc % Cc;
    const int hh = rem >> 6, dd = rem & 63;
#pragma unroll
    for (int r = 0; r < 8; ++r) {
      const int m = mbase + r;
      const int b = m >> 10, n = m & 1023;
      const size_t bhn = ((size_t)(b * Hh + hh) * Nn + n);
      const float v = acc[f][r];
      if (which == 0)      qh[(bhn << 6) + dd] = (_Float16)(v * 0.125f); // fold 1/sqrt(D)
      else if (which == 1) kh[(bhn << 6) + dd] = (_Float16)v;
      else vTh[(((size_t)(b * Hh + hh) << 6) + dd) * Nn + n] = (_Float16)v;
    }
  }
}

// ---------------- flash attention: per-(b,h), 128 Q rows per block ----------------
__global__ __launch_bounds__(256) void attn_kernel(const _Float16* __restrict__ qh,
                                                   const _Float16* __restrict__ kh,
                                                   const _Float16* __restrict__ vTh,
                                                   _Float16* __restrict__ aoh) {
  __shared__ __align__(16) _Float16 Ks[2][32 * 64];   // 32 keys x 64 d
  __shared__ __align__(16) _Float16 Vts[2][64 * 32];  // 64 d x 32 keys
  __shared__ __align__(16) _Float16 Ps[8][16 * 32];   // per-wave P tile

  const int bh = blockIdx.y;
  const int b = bh / Hh, h = bh % Hh;
  const int t = threadIdx.x;
  const int wave = t >> 5, lane = t & 31;
  const int q0 = blockIdx.x * 128 + wave * 16;

  const _Float16* qbase = qh + (size_t)bh * Nn * Dd;
  const _Float16* kbase = kh + (size_t)bh * Nn * Dd;
  const _Float16* vbase = vTh + (size_t)bh * Dd * Nn;

  // Q fragments live in registers for the whole kernel (scale pre-folded)
  v16h Qa0 = load_frag_a(qbase, Dd, q0, 0, lane);
  v16h Qa1 = load_frag_a(qbase, Dd, q0, 32, lane);

  v8f O[4];
  float mrow[8], lrow[8];
#pragma unroll
  for (int f = 0; f < 4; ++f)
#pragma unroll
    for (int r = 0; r < 8; ++r) O[f][r] = 0.0f;
#pragma unroll
  for (int r = 0; r < 8; ++r) { mrow[r] = -1e30f; lrow[r] = 0.0f; }

  _Float16* Pw = &Ps[wave][0];

  auto stage = [&](int buf, int jc) {
    int idx = t * 8;
    async_copy16(&Ks[buf][idx],
                 &kbase[(size_t)(jc + (idx >> 6)) * Dd + (idx & 63)]);
    async_copy16(&Vts[buf][idx],
                 &vbase[(size_t)(idx >> 5) * Nn + jc + (idx & 31)]);
  };

  stage(0, 0);
  wait_async0();
  __syncthreads();

  for (int i = 0; i < Nn / 32; ++i) {
    if (i + 1 < Nn / 32) stage((i + 1) & 1, (i + 1) * 32);
    const _Float16* Kb = Ks[i & 1];
    const _Float16* Vb = Vts[i & 1];

    // S = Q K^T  (16x32 per wave, two 16x16 C fragments)
    v8f S0, S1;
#pragma unroll
    for (int r = 0; r < 8; ++r) { S0[r] = 0.0f; S1[r] = 0.0f; }
    S0 = wmma_f16(Qa0, load_frag_b(Kb, 64, 0, 0, lane), S0);
    S1 = wmma_f16(Qa0, load_frag_b(Kb, 64, 16, 0, lane), S1);
    S0 = wmma_f16(Qa1, load_frag_b(Kb, 64, 0, 32, lane), S0);
    S1 = wmma_f16(Qa1, load_frag_b(Kb, 64, 16, 32, lane), S1);

    // online softmax; C-fragment row r -> matrix row r + 8*(lane>=16);
    // each matrix row occupies one 16-lane half -> reduce with xor 1,2,4,8
#pragma unroll
    for (int r = 0; r < 8; ++r) {
      float s0 = S0[r], s1 = S1[r];
      float mx = fmaxf(s0, s1);
#pragma unroll
      for (int off = 8; off >= 1; off >>= 1) mx = fmaxf(mx, __shfl_xor(mx, off, 32));
      const float mnew  = fmaxf(mrow[r], mx);
      const float alpha = __expf(mrow[r] - mnew);
      const float p0 = __expf(s0 - mnew);
      const float p1 = __expf(s1 - mnew);
      float rs = p0 + p1;
#pragma unroll
      for (int off = 8; off >= 1; off >>= 1) rs += __shfl_xor(rs, off, 32);
      lrow[r] = lrow[r] * alpha + rs;
      mrow[r] = mnew;
#pragma unroll
      for (int f = 0; f < 4; ++f) O[f][r] *= alpha;
      const int M = r + ((lane >> 4) << 3);
      const int nn = lane & 15;
      Pw[M * 32 + nn]      = (_Float16)p0;
      Pw[M * 32 + 16 + nn] = (_Float16)p1;
    }
    // same-wave LDS store->load: DS in-order per wave; drain DS counter
    wait_ds0();

    // O += P * V   (A = P 16x32, B fragments from V^T slab)
    v16h Pa = load_frag_a(Pw, 32, 0, 0, lane);
#pragma unroll
    for (int f = 0; f < 4; ++f)
      O[f] = wmma_f16(Pa, load_frag_b(Vb, 32, f * 16, 0, lane), O[f]);

    wait_async0();
    __syncthreads();
  }

  // normalize and write [B,N,C] f16 intermediate
#pragma unroll
  for (int f = 0; f < 4; ++f) {
#pragma unroll
    for (int r = 0; r < 8; ++r) {
      const int M = r + ((lane >> 4) << 3);
      const int n = q0 + M;
      const int dd = f * 16 + (lane & 15);
      const float val = O[f][r] / lrow[r];
      aoh[((size_t)(b * Nn + n)) * Cc + h * Dd + dd] = (_Float16)val;
    }
  }
}

// ---------------- projection GEMM: out = ao @ w_proj^T + b_proj ----------------
__global__ __launch_bounds__(256) void gemm_proj(const _Float16* __restrict__ aoh,
                                                 const _Float16* __restrict__ wh,
                                                 const float* __restrict__ bias,
                                                 float* __restrict__ out) {
  __shared__ __align__(16) _Float16 As[2][64 * 64];
  __shared__ __align__(16) _Float16 Bs[2][128 * 64];
  const int t = threadIdx.x;
  const int m0 = blockIdx.y * 64;
  const int n0 = blockIdx.x * 128;
  v8f acc[4];
  mm768x4(aoh, wh, m0, n0, As, Bs, t, acc);

  const int lane = t & 31, wave = t >> 5;
  const int wr = (wave >> 1) << 4, wc = (wave & 1) << 6;
  const int mbase = m0 + wr + ((lane >> 4) << 3);
#pragma unroll
  for (int f = 0; f < 4; ++f) {
    const int oc = n0 + wc + f * 16 + (lane & 15);
    const float bv = bias[oc];
#pragma unroll
    for (int r = 0; r < 8; ++r) {
      const int m = mbase + r;
      out[(size_t)m * Cc + oc] = acc[f][r] + bv;
    }
  }
}

// ---------------- launcher ----------------
extern "C" void kernel_launch(void* const* d_in, const int* in_sizes, int n_in,
                              void* d_out, int out_size, void* d_ws, size_t ws_size,
                              hipStream_t stream) {
  (void)in_sizes; (void)n_in; (void)out_size; (void)ws_size;
  const float* x      = (const float*)d_in[0];
  const float* w_qkv  = (const float*)d_in[1];
  const float* w_proj = (const float*)d_in[2];
  const float* b_proj = (const float*)d_in[3];
  float* out = (float*)d_out;

  _Float16* ws = (_Float16*)d_ws;
  _Float16* xh    = ws + OFF_XH;
  _Float16* wqkvh = ws + OFF_WQKV;
  _Float16* wprjh = ws + OFF_WPROJ;
  _Float16* qh    = ws + OFF_Q;
  _Float16* kh    = ws + OFF_K;
  _Float16* vTh   = ws + OFF_V;
  _Float16* aoh   = ws + OFF_AO;

  const int nx  = MTOT * Cc;
  const int nwq = QKV_O * Cc;
  const int nwp = Cc * Cc;
  cvt_f16<<<(nx / 8 + 255) / 256, 256, 0, stream>>>(x, xh, nx / 8);
  cvt_f16<<<(nwq / 8 + 255) / 256, 256, 0, stream>>>(w_qkv, wqkvh, nwq / 8);
  cvt_f16<<<(nwp / 8 + 255) / 256, 256, 0, stream>>>(w_proj, wprjh, nwp / 8);

  dim3 gq(QKV_O / 128, MTOT / 64);   // 18 x 128
  gemm_qkv<<<gq, 256, 0, stream>>>(xh, wqkvh, qh, kh, vTh);

  dim3 ga(Nn / 128, Bb * Hh);        // 8 x 96
  attn_kernel<<<ga, 256, 0, stream>>>(qh, kh, vTh, aoh);

  dim3 gp(Cc / 128, MTOT / 64);      // 6 x 128
  gemm_proj<<<gp, 256, 0, stream>>>(aoh, wprjh, b_proj, out);
}
